// StationGNN_75230647157397
// MI455X (gfx1250) — matrix operand order repaired and checked by reading, not compiled, not run on hardware
//
#include <hip/hip_runtime.h>
#include <hip/hip_bf16.h>

typedef __attribute__((ext_vector_type(16))) _Float16 v16h;
typedef __attribute__((ext_vector_type(8)))  float    v8f;

#define GN   50000
#define GE   800000
#define GH   128
#define GC   4
#define LDA  136   // 16x128 f16 tile row stride (pad 8) -> 68 dwords/row, conflict-free
#define LDM  72    // 16x64 f16 mid tile row stride

// ---------------- small utility kernels ----------------

__global__ void k_zero(float* __restrict__ p, size_t n) {
  size_t i = (size_t)blockIdx.x * blockDim.x + threadIdx.x;
  size_t stride = (size_t)gridDim.x * blockDim.x;
  for (; i < n; i += stride) p[i] = 0.0f;
}

__global__ void k_f32_to_f16(_Float16* __restrict__ dst, const float* __restrict__ src, int n) {
  int i = blockIdx.x * blockDim.x + threadIdx.x;
  if (i < n) dst[i] = (_Float16)src[i];
}

// Wh2 is (4,64); pad to (16,64) with zeros so the WMMA B operand needs no branches.
__global__ void k_pad_wh2(_Float16* __restrict__ dst, const float* __restrict__ src) {
  int i = blockIdx.x * blockDim.x + threadIdx.x;
  if (i < 16 * 64) {
    int r = i >> 6, c = i & 63;
    dst[i] = (r < GC) ? (_Float16)src[r * 64 + c] : (_Float16)0.0f;
  }
}

__global__ void k_degree(const int* __restrict__ ei, float* __restrict__ deg) {
  int e = blockIdx.x * blockDim.x + threadIdx.x;
  if (e < GE) atomicAdd(&deg[ei[GE + e]], 1.0f);
}

// one edge per 128 lanes: agg[dst][f] += x[src][f]
__global__ __launch_bounds__(256) void k_scatter(const int* __restrict__ ei,
                                                 const float* __restrict__ x,
                                                 float* __restrict__ agg) {
  int e = blockIdx.x * 2 + (threadIdx.x >> 7);
  int f = threadIdx.x & 127;
  if (e < GE) {
    int s = ei[e];
    int d = ei[GE + e];
    atomicAdd(&agg[(size_t)d * GH + f], x[(size_t)s * GH + f]);
  }
}

__global__ __launch_bounds__(256) void k_mean(float* __restrict__ agg,
                                              const float* __restrict__ deg) {
  size_t i = (size_t)blockIdx.x * blockDim.x + threadIdx.x;
  if (i < (size_t)GN * GH) {
    float d = deg[i >> 7];
    agg[i] = agg[i] / fmaxf(d, 1.0f);
  }
}

// ---------------- WMMA operand loaders ----------------
// A (16x32 f16, M x K): lane L holds row M = L&15; half = L>>4 selects K offset.
// VGPR v (element pair 2v,2v+1): K = 2v + 16*half  (+ kbase)
__device__ __forceinline__ v16h load_A(const _Float16* __restrict__ row, int kbase, int half) {
  union { v16h h; unsigned u[8]; } r;
#pragma unroll
  for (int v = 0; v < 8; ++v) {
    int k = 2 * v + 16 * half + kbase;
    r.u[v] = *reinterpret_cast<const unsigned*>(row + k);  // k even, row 16B-aligned
  }
  return r.h;
}

// B (32x16 f16, K x N): lane L holds column N = L&15; elements e=0..15 are
// K = kbase + 16*half + e. W row-major (fan_out, fan_in) => contiguous 32B run.
__device__ __forceinline__ v16h load_B(const _Float16* __restrict__ w, int fi,
                                       int col, int kbase, int half) {
  const uint4* p = reinterpret_cast<const uint4*>(w + (size_t)col * fi + kbase + 16 * half);
  union { v16h h; uint4 q[2]; } r;
  r.q[0] = p[0];
  r.q[1] = p[1];
  return r.h;
}

// ---------------- fused SAGE layer GEMM ----------------
// out[16 nodes x 128] = relu(mean @ Wl^T + bl + x @ Wr^T)
// 8 waves/block, one 16x16 output tile per wave, K=128 -> 4+4 WMMA steps.
__global__ __launch_bounds__(256) void k_sage_gemm(const float* __restrict__ mean,
                                                   const float* __restrict__ xroot,
                                                   const _Float16* __restrict__ WlH,
                                                   const _Float16* __restrict__ WrH,
                                                   const float* __restrict__ bl,
                                                   float* __restrict__ out) {
  __shared__ __align__(16) _Float16 sMean[16 * LDA];
  __shared__ __align__(16) _Float16 sX[16 * LDA];

  const int nb = blockIdx.x * 16;
  const int t  = threadIdx.x;

  for (int i = t; i < 16 * GH; i += 256) {
    int r = i >> 7, c = i & 127;
    sMean[r * LDA + c] = (_Float16)mean[(size_t)(nb + r) * GH + c];
    sX[r * LDA + c]    = (_Float16)xroot[(size_t)(nb + r) * GH + c];
  }
  __syncthreads();

  const int wave = t >> 5, lane = t & 31;
  const int half = lane >> 4, nIdx = lane & 15;
  const int colBase = wave * 16;

  v8f acc;
  const float b0 = bl[colBase + nIdx];
#pragma unroll
  for (int r = 0; r < 8; ++r) acc[r] = b0;

  const _Float16* mrow = &sMean[(lane & 15) * LDA];
  const _Float16* xrow = &sX[(lane & 15) * LDA];

#pragma unroll
  for (int kk = 0; kk < 4; ++kk) {
    v16h a = load_A(mrow, 32 * kk, half);
    v16h b = load_B(WlH, GH, colBase + nIdx, 32 * kk, half);
    acc = __builtin_amdgcn_wmma_f32_16x16x32_f16(false, a, false, b, (short)0, acc, false, false);
  }
#pragma unroll
  for (int kk = 0; kk < 4; ++kk) {
    v16h a = load_A(xrow, 32 * kk, half);
    v16h b = load_B(WrH, GH, colBase + nIdx, 32 * kk, half);
    acc = __builtin_amdgcn_wmma_f32_16x16x32_f16(false, a, false, b, (short)0, acc, false, false);
  }

  // C/D layout: lane -> N = nIdx; VGPR r -> M = r + 8*half
#pragma unroll
  for (int r = 0; r < 8; ++r) {
    int M = r + 8 * half;
    float v = acc[r];
    out[(size_t)(nb + M) * GH + colBase + nIdx] = v > 0.0f ? v : 0.0f;
  }
}

// ---------------- fused head MLP ----------------
// mid = relu(h @ Wh1^T + bh1)  (16x64, waves 0..3)
// out = mid @ Wh2^T + bh2      (16x4, wave 0)
__global__ __launch_bounds__(128) void k_head(const float* __restrict__ h,
                                              const _Float16* __restrict__ Wh1H,
                                              const float* __restrict__ bh1,
                                              const _Float16* __restrict__ Wh2H,  // padded 16x64
                                              const float* __restrict__ bh2,
                                              float* __restrict__ out) {
  __shared__ __align__(16) _Float16 sH[16 * LDA];
  __shared__ __align__(16) _Float16 sMid[16 * LDM];

  const int nb = blockIdx.x * 16;
  const int t  = threadIdx.x;

  for (int i = t; i < 16 * GH; i += 128) {
    int r = i >> 7, c = i & 127;
    sH[r * LDA + c] = (_Float16)h[(size_t)(nb + r) * GH + c];
  }
  __syncthreads();

  const int wave = t >> 5, lane = t & 31;
  const int half = lane >> 4, nIdx = lane & 15;

  {
    const int colBase = wave * 16;
    v8f acc;
    const float b0 = bh1[colBase + nIdx];
#pragma unroll
    for (int r = 0; r < 8; ++r) acc[r] = b0;
    const _Float16* arow = &sH[(lane & 15) * LDA];
#pragma unroll
    for (int kk = 0; kk < 4; ++kk) {
      v16h a = load_A(arow, 32 * kk, half);
      v16h b = load_B(Wh1H, GH, colBase + nIdx, 32 * kk, half);
      acc = __builtin_amdgcn_wmma_f32_16x16x32_f16(false, a, false, b, (short)0, acc, false, false);
    }
#pragma unroll
    for (int r = 0; r < 8; ++r) {
      int M = r + 8 * half;
      float v = acc[r];
      sMid[M * LDM + colBase + nIdx] = (_Float16)(v > 0.0f ? v : 0.0f);
    }
  }
  __syncthreads();

  if (wave == 0) {  // wave-uniform branch: EXEC all-1s inside
    v8f acc;
    const float b0 = (nIdx < GC) ? bh2[nIdx] : 0.0f;
#pragma unroll
    for (int r = 0; r < 8; ++r) acc[r] = b0;
    const _Float16* arow = &sMid[(lane & 15) * LDM];
#pragma unroll
    for (int kk = 0; kk < 2; ++kk) {
      v16h a = load_A(arow, 32 * kk, half);
      v16h b = load_B(Wh2H, 64, nIdx, 32 * kk, half);
      acc = __builtin_amdgcn_wmma_f32_16x16x32_f16(false, a, false, b, (short)0, acc, false, false);
    }
    if (nIdx < GC) {
#pragma unroll
      for (int r = 0; r < 8; ++r) {
        int M = r + 8 * half;
        out[(size_t)(nb + M) * GC + nIdx] = acc[r];
      }
    }
  }
}

// ---------------- host orchestration ----------------

extern "C" void kernel_launch(void* const* d_in, const int* in_sizes, int n_in,
                              void* d_out, int out_size, void* d_ws, size_t ws_size,
                              hipStream_t stream) {
  (void)in_sizes; (void)n_in; (void)out_size; (void)ws_size;

  const float* x   = (const float*)d_in[0];
  const int*   ei  = (const int*)d_in[1];
  const float* Wl[3] = {(const float*)d_in[2], (const float*)d_in[5], (const float*)d_in[8]};
  const float* bl[3] = {(const float*)d_in[3], (const float*)d_in[6], (const float*)d_in[9]};
  const float* Wr[3] = {(const float*)d_in[4], (const float*)d_in[7], (const float*)d_in[10]};
  const float* Wh1 = (const float*)d_in[11];
  const float* bh1 = (const float*)d_in[12];
  const float* Wh2 = (const float*)d_in[13];
  const float* bh2 = (const float*)d_in[14];
  float* out = (float*)d_out;

  // bump allocator over workspace (256B aligned slabs)
  char* wsp = (char*)d_ws;
  auto alloc = [&](size_t bytes) -> void* {
    void* p = (void*)wsp;
    wsp += (bytes + 255) & ~(size_t)255;
    return p;
  };
  float* deg = (float*)alloc((size_t)GN * sizeof(float));
  float* agg = (float*)alloc((size_t)GN * GH * sizeof(float));
  float* hA  = (float*)alloc((size_t)GN * GH * sizeof(float));
  float* hB  = (float*)alloc((size_t)GN * GH * sizeof(float));
  _Float16* WlH[3];
  _Float16* WrH[3];
  for (int l = 0; l < 3; ++l) {
    WlH[l] = (_Float16*)alloc((size_t)GH * GH * sizeof(_Float16));
    WrH[l] = (_Float16*)alloc((size_t)GH * GH * sizeof(_Float16));
  }
  _Float16* Wh1H = (_Float16*)alloc((size_t)64 * GH * sizeof(_Float16));
  _Float16* Wh2H = (_Float16*)alloc((size_t)16 * 64 * sizeof(_Float16));

  // weights -> f16 (once per launch)
  for (int l = 0; l < 3; ++l) {
    k_f32_to_f16<<<(GH * GH + 255) / 256, 256, 0, stream>>>(WlH[l], Wl[l], GH * GH);
    k_f32_to_f16<<<(GH * GH + 255) / 256, 256, 0, stream>>>(WrH[l], Wr[l], GH * GH);
  }
  k_f32_to_f16<<<(64 * GH + 255) / 256, 256, 0, stream>>>(Wh1H, Wh1, 64 * GH);
  k_pad_wh2<<<(16 * 64 + 255) / 256, 256, 0, stream>>>(Wh2H, Wh2);

  // degree (edges fixed across layers)
  k_zero<<<256, 256, 0, stream>>>(deg, (size_t)GN);
  k_degree<<<(GE + 255) / 256, 256, 0, stream>>>(ei, deg);

  const float* cur = x;
  float* bufs[2] = {hA, hB};
  for (int l = 0; l < 3; ++l) {
    k_zero<<<4096, 256, 0, stream>>>(agg, (size_t)GN * GH);
    k_scatter<<<(GE + 1) / 2, 256, 0, stream>>>(ei, cur, agg);
    k_mean<<<(GN * GH + 255) / 256, 256, 0, stream>>>(agg, deg);
    float* nxt = bufs[l & 1];
    k_sage_gemm<<<GN / 16, 256, 0, stream>>>(agg, cur, WlH[l], WrH[l], bl[l], nxt);
    cur = nxt;
  }

  k_head<<<GN / 16, 128, 0, stream>>>(cur, Wh1H, bh1, Wh2H, bh2, out);
}
